// MoEAudioProjector_32607391711812
// MI455X (gfx1250) — matrix operand order repaired and checked by reading, not compiled, not run on hardware
//
#include <hip/hip_runtime.h>
#include <hip/hip_bf16.h>

// ---------------------------------------------------------------------------
// MoE audio projector for gfx1250 (MI455X).
// bf16 WMMA GEMMs (v_wmma_f32_16x16x32_bf16), one-shot fp32->bf16 weight
// conversion, token gathering for routed experts (top-2 of 8), fused SwiGLU,
// double-buffered LDS staging (1 barrier / K-step) + global prefetch.
// ---------------------------------------------------------------------------

typedef __bf16 bf16_t;
typedef __bf16 bf16x16 __attribute__((ext_vector_type(16)));
typedef float  f32x8   __attribute__((ext_vector_type(8)));

struct __attribute__((aligned(16))) U4 { unsigned x, y, z, w; };
union FragU { U4 q[2]; bf16x16 v; };
struct __attribute__((aligned(8))) B4 { bf16_t a, b, c, d; };

#define T_TOK   12000
#define DIM_IN  2048
#define DIM_H   2048
#define DIM_O   2048
#define NEXP    8
#define MT      128
#define NT      64
#define KT      32
#define LDK     40   // padded LDS row stride in halfs (80 B, 16B aligned)

// ----------------------- fp32 -> bf16 weight convert -----------------------
__global__ void cvt_bf16_k(const float* __restrict__ src,
                           bf16_t* __restrict__ dst, long n)
{
  const long stride = (long)gridDim.x * blockDim.x * 4;
  for (long i = ((long)blockIdx.x * blockDim.x + threadIdx.x) * 4; i < n;
       i += stride) {
    const float* s = src + i;
    B4 o;
    o.a = (bf16_t)s[0]; o.b = (bf16_t)s[1];
    o.c = (bf16_t)s[2]; o.d = (bf16_t)s[3];
    *(B4*)(dst + i) = o;
  }
}

// --------------------------- pool + pre-RMSNorm ----------------------------
__global__ void pool_rmsnorm_k(const float* __restrict__ x,
                               const float* __restrict__ lnw,
                               bf16_t* __restrict__ normx,
                               float* __restrict__ nxn)
{
  const int t = blockIdx.x;
  const float* xr = x + (size_t)t * DIM_IN;
  __shared__ float r1[256], r2[256];
  float v[8], w[8];
  float ss = 0.f, sw = 0.f;
#pragma unroll
  for (int i = 0; i < 8; ++i) {
    int c = threadIdx.x + i * 256;
    v[i] = xr[c]; w[i] = lnw[c];
    ss += v[i] * v[i];
    float z = v[i] * w[i];
    sw += z * z;
  }
  r1[threadIdx.x] = ss; r2[threadIdx.x] = sw;
  __syncthreads();
  for (int s = 128; s > 0; s >>= 1) {
    if (threadIdx.x < s) {
      r1[threadIdx.x] += r1[threadIdx.x + s];
      r2[threadIdx.x] += r2[threadIdx.x + s];
    }
    __syncthreads();
  }
  const float scale = rsqrtf(r1[0] * (1.0f / DIM_IN) + 1e-6f);
#pragma unroll
  for (int i = 0; i < 8; ++i) {
    int c = threadIdx.x + i * 256;
    normx[(size_t)t * DIM_IN + c] = (bf16_t)(v[i] * w[i] * scale);
  }
  if (threadIdx.x == 0) nxn[t] = sqrtf(r2[0]) * scale;
}

// --------------------------- router weight norms ---------------------------
__global__ void rwnorm_k(const float* __restrict__ rw, float* __restrict__ rwn)
{
  const int e = blockIdx.x;
  __shared__ float red[256];
  float s = 0.f;
  for (int k = threadIdx.x; k < DIM_IN; k += 256) {
    float v = rw[(size_t)e * DIM_IN + k];
    s += v * v;
  }
  red[threadIdx.x] = s; __syncthreads();
  for (int q = 128; q > 0; q >>= 1) {
    if (threadIdx.x < q) red[threadIdx.x] += red[threadIdx.x + q];
    __syncthreads();
  }
  if (threadIdx.x == 0) rwn[e] = sqrtf(red[0]);
}

// ------------------------------- router ------------------------------------
__global__ void router_k(const bf16_t* __restrict__ normx,
                         const float* __restrict__ rw,
                         const float* __restrict__ rwn,
                         const float* __restrict__ nxn,
                         const float* __restrict__ eload,
                         int* __restrict__ counts,
                         int* __restrict__ tope,
                         float* __restrict__ topw)
{
  const int lane = threadIdx.x & 31;
  const int wid  = threadIdx.x >> 5;
  const int t = blockIdx.x * 8 + wid;
  if (t >= T_TOK) return;
  float acc[NEXP];
#pragma unroll
  for (int e = 0; e < NEXP; ++e) acc[e] = 0.f;
  const bf16_t* xr = normx + (size_t)t * DIM_IN;
  for (int k = lane; k < DIM_IN; k += 32) {
    const float xv = (float)xr[k];
#pragma unroll
    for (int e = 0; e < NEXP; ++e) acc[e] += xv * rw[(size_t)e * DIM_IN + k];
  }
#pragma unroll
  for (int e = 0; e < NEXP; ++e)
    for (int off = 16; off > 0; off >>= 1)
      acc[e] += __shfl_xor(acc[e], off, 32);
  if (lane == 0) {
    const float nxv = nxn[t];
    float p[NEXP];
    int b0 = 0, b1 = 0;
    float c0 = -1e30f, c1 = -1e30f;
#pragma unroll
    for (int e = 0; e < NEXP; ++e) {
      float logit = acc[e] / (nxv * rwn[e]) * 16.0f;
      p[e] = 1.0f / (1.0f + __expf(-logit));
      float ch = p[e] - 0.5f * eload[e];
      if (ch > c0) { c1 = c0; b1 = b0; c0 = ch; b0 = e; }
      else if (ch > c1) { c1 = ch; b1 = e; }
    }
    const float wsum = p[b0] + p[b1] + 1e-20f;
    tope[2 * t]     = b0;
    tope[2 * t + 1] = b1;
    topw[2 * t]     = p[b0] / wsum;   // ROUTED_SCALING = 1.0
    topw[2 * t + 1] = p[b1] / wsum;
    atomicAdd(&counts[b0], 1);
    atomicAdd(&counts[b1], 1);
  }
}

// ------------------------ scan + compaction build --------------------------
__global__ void scan_k(const int* __restrict__ counts,
                       int* __restrict__ offs,
                       int* __restrict__ cursors)
{
  if (threadIdx.x == 0) {
    int a = 0;
    for (int e = 0; e < NEXP; ++e) { offs[e] = a; a += counts[e]; }
    offs[NEXP] = a;
  }
  if (threadIdx.x < NEXP) cursors[threadIdx.x] = 0;
}

__global__ void build_k(const int* __restrict__ tope,
                        int* __restrict__ cursors,
                        const int* __restrict__ offs,
                        int* __restrict__ gidx,
                        int* __restrict__ posb)
{
  const int t = blockIdx.x * 256 + threadIdx.x;
  if (t >= T_TOK) return;
#pragma unroll
  for (int s = 0; s < 2; ++s) {
    const int e = tope[2 * t + s];
    const int pos = atomicAdd(&cursors[e], 1);
    const int g = offs[e] + pos;
    gidx[g] = t;
    posb[2 * t + s] = g;
  }
}

// ------------------- GEMM1 (x @ w12^T) fused with SwiGLU -------------------
// 256 threads = 8 waves; tile 128(M) x 64(N); K step 32; double-buffered LDS.
// B tile holds 128 rows: [0,64) = gate rows, [64,128) = val rows.
__global__ void gemm1_swiglu_k(const bf16_t* __restrict__ A,   // [T, DIM_IN]
                               const bf16_t* __restrict__ W12, // [(E,)2H, DIM_IN] bf16
                               bf16_t* __restrict__ inter,     // [rows, DIM_H]
                               const int* __restrict__ gidx,   // compact->token or null
                               const int* __restrict__ offs)   // [E+1] or null
{
  int base = 0, nrows = T_TOK;
  const bf16_t* Wp = W12;
  if (offs) {
    const int e = blockIdx.z;
    base  = offs[e];
    nrows = offs[e + 1] - base;
    Wp    = W12 + (size_t)e * (2 * DIM_H) * DIM_IN;
  }
  const int m0 = blockIdx.x * MT;
  if (m0 >= nrows) return;
  const int n0 = blockIdx.y * NT;

  __shared__ bf16_t As[2][MT][LDK];
  __shared__ bf16_t Bs[2][MT][LDK];   // gate | val

  const int tid   = threadIdx.x;
  const int lane  = tid & 31;
  const int wid   = tid >> 5;
  const int wm    = wid & 3;    // 4 M-subtiles of 32
  const int wn    = wid >> 2;   // 2 N-subtiles of 32
  const int khalf = lane >> 4;
  const int l16   = lane & 15;

  f32x8 zerov;
#pragma unroll
  for (int i = 0; i < 8; ++i) zerov[i] = 0.f;
  f32x8 cg[2][2], cv[2][2];
#pragma unroll
  for (int mf = 0; mf < 2; ++mf)
#pragma unroll
    for (int nf = 0; nf < 2; ++nf) { cg[mf][nf] = zerov; cv[mf][nf] = zerov; }

  // staging map: 2 threads / row, 32 B each
  const int srow  = tid & 127;
  const int spart = tid >> 7;
  const bf16_t* arp = nullptr;
  if (m0 + srow < nrows) {
    const int tok = gidx ? gidx[base + m0 + srow] : (m0 + srow);
    arp = A + (size_t)tok * DIM_IN;
  }
  const int wrow = (srow < 64) ? (n0 + srow) : (DIM_H + n0 + (srow - 64));
  const bf16_t* brp = Wp + (size_t)wrow * DIM_IN;

  auto stage = [&](int b, int k0) {
    const int h0 = spart * 16;
    if (arp) {
      *(U4*)&As[b][srow][h0]     = *(const U4*)(arp + k0 + h0);
      *(U4*)&As[b][srow][h0 + 8] = *(const U4*)(arp + k0 + h0 + 8);
    } else {
      U4 z4 = {0, 0, 0, 0};
      *(U4*)&As[b][srow][h0] = z4;
      *(U4*)&As[b][srow][h0 + 8] = z4;
    }
    *(U4*)&Bs[b][srow][h0]     = *(const U4*)(brp + k0 + h0);
    *(U4*)&Bs[b][srow][h0 + 8] = *(const U4*)(brp + k0 + h0 + 8);
    if (k0 + 2 * KT < DIM_IN) {               // prefetch tile k0+64
      if (arp) __builtin_prefetch(arp + k0 + 2 * KT + h0, 0, 0);
      __builtin_prefetch(brp + k0 + 2 * KT + h0, 0, 0);
    }
  };

  stage(0, 0);
  __syncthreads();

  const int nk = DIM_IN / KT;
  for (int it = 0; it < nk; ++it) {
    const int buf = it & 1;
    if (it + 1 < nk) stage(buf ^ 1, (it + 1) * KT);

    bf16x16 af[2], bgf[2], bvf[2];
#pragma unroll
    for (int mf = 0; mf < 2; ++mf) {
      const int r = wm * 32 + mf * 16 + l16;
      FragU u;
      u.q[0] = *(const U4*)&As[buf][r][16 * khalf];
      u.q[1] = *(const U4*)&As[buf][r][16 * khalf + 8];
      af[mf] = u.v;
    }
#pragma unroll
    for (int nf = 0; nf < 2; ++nf) {
      const int n = wn * 32 + nf * 16 + l16;
      FragU u, u2;
      u.q[0]  = *(const U4*)&Bs[buf][n][16 * khalf];
      u.q[1]  = *(const U4*)&Bs[buf][n][16 * khalf + 8];
      bgf[nf] = u.v;
      u2.q[0] = *(const U4*)&Bs[buf][64 + n][16 * khalf];
      u2.q[1] = *(const U4*)&Bs[buf][64 + n][16 * khalf + 8];
      bvf[nf] = u2.v;
    }
#pragma unroll
    for (int mf = 0; mf < 2; ++mf)
#pragma unroll
      for (int nf = 0; nf < 2; ++nf) {
        cg[mf][nf] = __builtin_amdgcn_wmma_f32_16x16x32_bf16(
            false, af[mf], false, bgf[nf], (short)0, cg[mf][nf], false, false);
        cv[mf][nf] = __builtin_amdgcn_wmma_f32_16x16x32_bf16(
            false, af[mf], false, bvf[nf], (short)0, cv[mf][nf], false, false);
      }
    __syncthreads();
  }

  // SwiGLU epilogue: silu(gate) * val -> bf16 intermediate
#pragma unroll
  for (int mf = 0; mf < 2; ++mf)
#pragma unroll
    for (int nf = 0; nf < 2; ++nf)
#pragma unroll
      for (int r = 0; r < 8; ++r) {
        const int rl = m0 + wm * 32 + mf * 16 + r + 8 * khalf;
        if (rl < nrows) {
          const int col = n0 + wn * 32 + nf * 16 + l16;
          const float g = cg[mf][nf][r];
          const float vv = cv[mf][nf][r];
          const float h = (g / (1.0f + __expf(-g))) * vv;
          inter[(size_t)(base + rl) * DIM_H + col] = (bf16_t)h;
        }
      }
}

// ------------------------- GEMM2 (h @ w3^T) -> f32 -------------------------
__global__ void gemm2_k(const bf16_t* __restrict__ A,  // [rows, DIM_H] (compact)
                        const bf16_t* __restrict__ W3, // [(E,)DIM_O, DIM_H] bf16
                        float* __restrict__ outp,      // [rows, DIM_O]
                        const int* __restrict__ offs)
{
  int base = 0, nrows = T_TOK;
  const bf16_t* Wp = W3;
  if (offs) {
    const int e = blockIdx.z;
    base  = offs[e];
    nrows = offs[e + 1] - base;
    Wp    = W3 + (size_t)e * DIM_O * DIM_H;
  }
  const int m0 = blockIdx.x * MT;
  if (m0 >= nrows) return;
  const int n0 = blockIdx.y * NT;

  __shared__ bf16_t As[2][MT][LDK];
  __shared__ bf16_t Bs[2][NT][LDK];

  const int tid   = threadIdx.x;
  const int lane  = tid & 31;
  const int wid   = tid >> 5;
  const int wm    = wid & 3;
  const int wn    = wid >> 2;
  const int khalf = lane >> 4;
  const int l16   = lane & 15;

  f32x8 zerov;
#pragma unroll
  for (int i = 0; i < 8; ++i) zerov[i] = 0.f;
  f32x8 c[2][2];
#pragma unroll
  for (int mf = 0; mf < 2; ++mf)
#pragma unroll
    for (int nf = 0; nf < 2; ++nf) c[mf][nf] = zerov;

  const int arow  = tid & 127;
  const int apart = tid >> 7;
  const bf16_t* arp =
      (m0 + arow < nrows) ? A + (size_t)(base + m0 + arow) * DIM_H : nullptr;
  const int brow  = tid & 63;
  const int bpart = tid >> 6;           // 0..3, 16 B each
  const bf16_t* brp = Wp + (size_t)(n0 + brow) * DIM_H;

  auto stage = [&](int b, int k0) {
    const int h0 = apart * 16;
    if (arp) {
      *(U4*)&As[b][arow][h0]     = *(const U4*)(arp + k0 + h0);
      *(U4*)&As[b][arow][h0 + 8] = *(const U4*)(arp + k0 + h0 + 8);
    } else {
      U4 z4 = {0, 0, 0, 0};
      *(U4*)&As[b][arow][h0] = z4;
      *(U4*)&As[b][arow][h0 + 8] = z4;
    }
    *(U4*)&Bs[b][brow][bpart * 8] = *(const U4*)(brp + k0 + bpart * 8);
    if (k0 + 2 * KT < DIM_H) {
      if (arp) __builtin_prefetch(arp + k0 + 2 * KT + h0, 0, 0);
      __builtin_prefetch(brp + k0 + 2 * KT + bpart * 8, 0, 0);
    }
  };

  stage(0, 0);
  __syncthreads();

  const int nk = DIM_H / KT;
  for (int it = 0; it < nk; ++it) {
    const int buf = it & 1;
    if (it + 1 < nk) stage(buf ^ 1, (it + 1) * KT);

    bf16x16 af[2], bf_[2];
#pragma unroll
    for (int mf = 0; mf < 2; ++mf) {
      const int r = wm * 32 + mf * 16 + l16;
      FragU u;
      u.q[0] = *(const U4*)&As[buf][r][16 * khalf];
      u.q[1] = *(const U4*)&As[buf][r][16 * khalf + 8];
      af[mf] = u.v;
    }
#pragma unroll
    for (int nf = 0; nf < 2; ++nf) {
      const int n = wn * 32 + nf * 16 + l16;
      FragU u;
      u.q[0]  = *(const U4*)&Bs[buf][n][16 * khalf];
      u.q[1]  = *(const U4*)&Bs[buf][n][16 * khalf + 8];
      bf_[nf] = u.v;
    }
#pragma unroll
    for (int mf = 0; mf < 2; ++mf)
#pragma unroll
      for (int nf = 0; nf < 2; ++nf)
        c[mf][nf] = __builtin_amdgcn_wmma_f32_16x16x32_bf16(
            false, af[mf], false, bf_[nf], (short)0, c[mf][nf], false, false);
    __syncthreads();
  }

#pragma unroll
  for (int mf = 0; mf < 2; ++mf)
#pragma unroll
    for (int nf = 0; nf < 2; ++nf)
#pragma unroll
      for (int r = 0; r < 8; ++r) {
        const int rl = m0 + wm * 32 + mf * 16 + r + 8 * khalf;
        if (rl < nrows) {
          const int col = n0 + wn * 32 + nf * 16 + l16;
          outp[(size_t)(base + rl) * DIM_O + col] = c[mf][nf][r];
        }
      }
}

// -------------------- combine + post-RMSNorm -> output ---------------------
__global__ void final_k(const float* __restrict__ so,
                        const float* __restrict__ yr,
                        const int* __restrict__ posb,
                        const float* __restrict__ topw,
                        const float* __restrict__ lnpost,
                        float* __restrict__ outp)
{
  const int t = blockIdx.x;
  const int p0 = posb[2 * t], p1 = posb[2 * t + 1];
  const float w0 = topw[2 * t], w1 = topw[2 * t + 1];
  __shared__ float red[256];
  float v[8];
  float ss = 0.f;
#pragma unroll
  for (int i = 0; i < 8; ++i) {
    const int cidx = threadIdx.x + i * 256;
    const float s = so[(size_t)t * DIM_O + cidx]
                  + w0 * yr[(size_t)p0 * DIM_O + cidx]
                  + w1 * yr[(size_t)p1 * DIM_O + cidx];
    v[i] = s; ss += s * s;
  }
  red[threadIdx.x] = ss; __syncthreads();
  for (int s = 128; s > 0; s >>= 1) {
    if (threadIdx.x < s) red[threadIdx.x] += red[threadIdx.x + s];
    __syncthreads();
  }
  const float scale = rsqrtf(red[0] * (1.0f / DIM_O) + 1e-6f);
#pragma unroll
  for (int i = 0; i < 8; ++i) {
    const int cidx = threadIdx.x + i * 256;
    outp[(size_t)t * DIM_O + cidx] = v[i] * scale * lnpost[cidx];
  }
}

// ---------------------------------------------------------------------------
extern "C" void kernel_launch(void* const* d_in, const int* in_sizes, int n_in,
                              void* d_out, int out_size, void* d_ws, size_t ws_size,
                              hipStream_t stream)
{
  const float* x        = (const float*)d_in[0];
  const float* ln_pre   = (const float*)d_in[1];
  const float* ln_post  = (const float*)d_in[2];
  const float* router_w = (const float*)d_in[3];
  const float* sh_w12   = (const float*)d_in[4];
  const float* sh_w3    = (const float*)d_in[5];
  const float* ex_w12   = (const float*)d_in[6];
  const float* ex_w3    = (const float*)d_in[7];
  const float* ex_load  = (const float*)d_in[8];
  float* out = (float*)d_out;

  char* ws = (char*)d_ws;
  size_t cur = 0;
  auto carve = [&](size_t bytes) -> char* {
    char* p = ws + cur;
    cur = (cur + bytes + 255) & ~(size_t)255;
    return p;
  };
  bf16_t* normx  = (bf16_t*)carve((size_t)T_TOK * DIM_IN * 2);
  float*  nxn    = (float*)carve((size_t)T_TOK * 4);
  float*  rwn    = (float*)carve(64);
  int*    tope   = (int*)carve((size_t)2 * T_TOK * 4);
  float*  topw   = (float*)carve((size_t)2 * T_TOK * 4);
  int*    counts = (int*)carve(64);
  int*    offs   = (int*)carve(64);
  int*    cursors= (int*)carve(64);
  int*    gidx   = (int*)carve((size_t)2 * T_TOK * 4);
  int*    posb   = (int*)carve((size_t)2 * T_TOK * 4);
  bf16_t* interS = (bf16_t*)carve((size_t)T_TOK * DIM_H * 2);
  bf16_t* interR = (bf16_t*)carve((size_t)2 * T_TOK * DIM_H * 2);
  float*  shout  = (float*)carve((size_t)T_TOK * DIM_O * 4);
  float*  yrout  = (float*)carve((size_t)2 * T_TOK * DIM_O * 4);
  // bf16 weight mirrors
  const long nShW12 = (long)2 * DIM_H * DIM_IN;
  const long nShW3  = (long)DIM_O * DIM_H;
  const long nExW12 = (long)NEXP * 2 * DIM_H * DIM_IN;
  const long nExW3  = (long)NEXP * DIM_O * DIM_H;
  bf16_t* shW12b = (bf16_t*)carve((size_t)nShW12 * 2);
  bf16_t* shW3b  = (bf16_t*)carve((size_t)nShW3 * 2);
  bf16_t* exW12b = (bf16_t*)carve((size_t)nExW12 * 2);
  bf16_t* exW3b  = (bf16_t*)carve((size_t)nExW3 * 2);
  (void)ws_size; (void)in_sizes; (void)n_in; (void)out_size;

  hipMemsetAsync(counts, 0, 64, stream);

  cvt_bf16_k<<<2048, 256, 0, stream>>>(sh_w12, shW12b, nShW12);
  cvt_bf16_k<<<2048, 256, 0, stream>>>(sh_w3,  shW3b,  nShW3);
  cvt_bf16_k<<<4096, 256, 0, stream>>>(ex_w12, exW12b, nExW12);
  cvt_bf16_k<<<4096, 256, 0, stream>>>(ex_w3,  exW3b,  nExW3);

  rwnorm_k<<<NEXP, 256, 0, stream>>>(router_w, rwn);
  pool_rmsnorm_k<<<T_TOK, 256, 0, stream>>>(x, ln_pre, normx, nxn);
  router_k<<<(T_TOK + 7) / 8, 256, 0, stream>>>(normx, router_w, rwn, nxn,
                                                ex_load, counts, tope, topw);
  scan_k<<<1, 64, 0, stream>>>(counts, offs, cursors);
  build_k<<<(T_TOK + 255) / 256, 256, 0, stream>>>(tope, cursors, offs, gidx, posb);

  const int mtiles = (T_TOK + MT - 1) / MT;   // 94
  dim3 gs1(mtiles, DIM_H / NT);               // shared expert
  gemm1_swiglu_k<<<gs1, 256, 0, stream>>>(normx, shW12b, interS, nullptr, nullptr);
  dim3 gs2(mtiles, DIM_O / NT);
  gemm2_k<<<gs2, 256, 0, stream>>>(interS, shW3b, shout, nullptr);

  dim3 gr1(mtiles, DIM_H / NT, NEXP);         // routed experts (gathered rows)
  gemm1_swiglu_k<<<gr1, 256, 0, stream>>>(normx, exW12b, interR, gidx, offs);
  dim3 gr2(mtiles, DIM_O / NT, NEXP);
  gemm2_k<<<gr2, 256, 0, stream>>>(interR, exW3b, yrout, offs);

  final_k<<<T_TOK, 256, 0, stream>>>(shout, yrout, posb, topw, ln_post, out);

  // aux_loss (second tuple element) = 0.0f, appended after the [T, O] tensor
  hipMemsetAsync((char*)out + (size_t)T_TOK * DIM_O * 4, 0, 4, stream);
}
